// Attention_16973710754318
// MI455X (gfx1250) — matrix operand order repaired and checked by reading, not compiled, not run on hardware
//
#include <hip/hip_runtime.h>
#include <hip/hip_bf16.h>
#include <cstddef>
#include <cstdint>
#include <type_traits>

// ---------------------------------------------------------------------------
// Problem constants (from the reference)
// ---------------------------------------------------------------------------
constexpr int B_   = 2;
constexpr int S_   = 2048;
constexpr int D_   = 2048;
constexpr int H_   = 16;
constexpr int KVH_ = 4;
constexpr int HD_  = 128;           // head dim
constexpr int KVD_ = KVH_ * HD_;    // 512
constexpr float EPS_ = 1.1920929e-07f;
// inv_freq[i] = exp(-i * ln(10000)/64)
constexpr float LN1E4_OVER_64 = 0.14391156509f;

typedef __attribute__((ext_vector_type(16))) __bf16 v16bf;
typedef __attribute__((ext_vector_type(8)))  float  v8f;

__device__ __forceinline__ v8f wmma_bf16(v16bf a, v16bf b, v8f c) {
  // (neg_a, A, neg_b, B, c_mod, C, reuse_a, reuse_b)
  return __builtin_amdgcn_wmma_f32_16x16x32_bf16(false, a, false, b,
                                                 (short)0, c, false, false);
}

__device__ __forceinline__ __bf16 to_bf16(float x) { return (__bf16)x; }

// Contiguous 32-byte fragment load from LDS -> 2x ds_load_b128
__device__ __forceinline__ v16bf ld_frag(const __bf16* p) {
  return *reinterpret_cast<const v16bf*>(p);
}

// ---------------------------------------------------------------------------
// CDNA5 async global->LDS copy (ASYNCcnt-tracked, bypasses VGPRs).
// Per ISA 10.2: generic LDS addresses carry the LDS byte offset in the low
// 32 bits, so truncation yields the VDST operand.
// ---------------------------------------------------------------------------
__device__ __forceinline__ void async_ld_b128(void* lds_ptr, const void* gptr) {
  const uint32_t lds_off = (uint32_t)(uintptr_t)lds_ptr;
  asm volatile("global_load_async_to_lds_b128 %0, %1, off"
               :: "v"(lds_off), "v"(gptr)
               : "memory");
}
__device__ __forceinline__ void wait_async0() {
  asm volatile("s_wait_asynccnt 0" ::: "memory");
}

// A-side K permutation sigma: swap 8-blocks [8,16) <-> [16,24) within each 32.
// (WMMA pairs operands by hardware K slot; contiguous per-lane reads realize
//  the same k = 16h + s order for A and B, so no permutation is needed; A is
//  staged naturally and these helpers are unused.)
__device__ __forceinline__ int swz_k(int k) {
  const int blk = k & 24;
  return (blk == 8) ? k + 8 : (blk == 16) ? k - 8 : k;
}
// same permutation on aligned groups of 8 (group index within a 32-block)
__device__ __forceinline__ int swz_g(int g) {
  const int gl = g & 3;
  return (g & ~3) | ((gl == 1) ? 2 : (gl == 2) ? 1 : gl);
}

// ---------------------------------------------------------------------------
// Generic GEMM:  C[m][n] = sum_k A[m][k] * Bm[n][k]  + bias[n]
// 64x64 block tile, BK=32, 256 threads = 8 waves, each wave -> two 16x16 frags
// ---------------------------------------------------------------------------
template <typename AT>
__global__ __launch_bounds__(256) void gemm_abt_wmma(
    const AT* __restrict__ A, int lda,
    const float* __restrict__ Bm, int ldb,
    const float* __restrict__ bias,
    float* __restrict__ C, int ldc, int K)
{
  __shared__ __bf16 As[64][32];   // A natural
  __shared__ __bf16 Bs[64][32];   // B natural

  const int tid  = threadIdx.x;
  const int w    = tid >> 5;       // wave id 0..7 (wave32)
  const int lane = tid & 31;
  const int m0 = blockIdx.y * 64;
  const int n0 = blockIdx.x * 64;
  const int wm = (w & 3) * 16;
  const int wn = (w >> 2) * 32;

  v8f acc0 = {}, acc1 = {};

  const int lr  = tid >> 2;            // tile row to stage
  const int lc  = (tid & 3) * 8;       // 8 consecutive K elems (one group)
  const int lcA = lc;                  // natural destination group

  const int lrow  = lane & 15;
  const int halfo = (lane >> 4) << 4;  // 0 or 16 (32B-aligned frag offset)

  for (int k0 = 0; k0 < K; k0 += 32) {
    // ---- stage tiles into LDS --------------------------------------------
    {
      __bf16 tb[8];
      if constexpr (std::is_same<AT, float>::value) {
        __bf16 ta[8];
        const float4 a0 = *reinterpret_cast<const float4*>(
            &A[(size_t)(m0 + lr) * lda + k0 + lc]);
        const float4 a1 = *reinterpret_cast<const float4*>(
            &A[(size_t)(m0 + lr) * lda + k0 + lc + 4]);
        ta[0] = to_bf16(a0.x); ta[1] = to_bf16(a0.y);
        ta[2] = to_bf16(a0.z); ta[3] = to_bf16(a0.w);
        ta[4] = to_bf16(a1.x); ta[5] = to_bf16(a1.y);
        ta[6] = to_bf16(a1.z); ta[7] = to_bf16(a1.w);
        *reinterpret_cast<uint4*>(&As[lr][lcA]) = *reinterpret_cast<uint4*>(ta);
      } else {
        // bf16 A: pure copy -> CDNA5 async global->LDS (no VGPR round trip)
        async_ld_b128(&As[lr][lcA], &A[(size_t)(m0 + lr) * lda + k0 + lc]);
      }
      const float4 b0v = *reinterpret_cast<const float4*>(
          &Bm[(size_t)(n0 + lr) * ldb + k0 + lc]);
      const float4 b1v = *reinterpret_cast<const float4*>(
          &Bm[(size_t)(n0 + lr) * ldb + k0 + lc + 4]);
      tb[0] = to_bf16(b0v.x); tb[1] = to_bf16(b0v.y);
      tb[2] = to_bf16(b0v.z); tb[3] = to_bf16(b0v.w);
      tb[4] = to_bf16(b1v.x); tb[5] = to_bf16(b1v.y);
      tb[6] = to_bf16(b1v.z); tb[7] = to_bf16(b1v.w);
      *reinterpret_cast<uint4*>(&Bs[lr][lc]) = *reinterpret_cast<uint4*>(tb);
    }
    // prefetch next K tile (global_prefetch_b8)
    if (k0 + 32 < K) {
      __builtin_prefetch(&A[(size_t)(m0 + lr) * lda + k0 + 32 + lc], 0, 1);
      __builtin_prefetch(&Bm[(size_t)(n0 + lr) * ldb + k0 + 32 + lc], 0, 1);
    }
    if constexpr (!std::is_same<AT, float>::value) wait_async0();
    __syncthreads();

    // ---- fragments: contiguous 32B loads (2x ds_load_b128 each) ----------
    const v16bf af = ld_frag(&As[wm + lrow][halfo]);
    const v16bf b0 = ld_frag(&Bs[wn + lrow][halfo]);
    const v16bf b1 = ld_frag(&Bs[wn + 16 + lrow][halfo]);
    acc0 = wmma_bf16(af, b0, acc0);
    acc1 = wmma_bf16(af, b1, acc1);
    __syncthreads();
  }

  // ---- epilogue: C layout (vgpr r -> M=r(+8 for hi lanes), N=lane) -------
  const int col = lane & 15;
  const int rb  = (lane >> 4) << 3;
  #pragma unroll
  for (int r = 0; r < 8; ++r) {
    int m = m0 + wm + rb + r;
    C[(size_t)m * ldc + n0 + wn + col]      = acc0[r] + bias[n0 + wn + col];
    C[(size_t)m * ldc + n0 + wn + 16 + col] = acc1[r] + bias[n0 + wn + 16 + col];
  }
}

// ---------------------------------------------------------------------------
// RMSNorm + RoPE (+ optional gain*scale), fp32 [B,S,NH,HD] -> bf16 [B,NH,S,HD]
// ---------------------------------------------------------------------------
__global__ __launch_bounds__(128) void rmsnorm_rope(
    const float* __restrict__ src, __bf16* __restrict__ dst,
    const float* __restrict__ gain, float scale)
{
  const int s  = blockIdx.x;
  const int h  = blockIdx.y;
  const int b  = blockIdx.z;
  const int NH = gridDim.y;
  const int d  = threadIdx.x;

  __shared__ float red[128];
  __shared__ float nv[128];

  const size_t si = (((size_t)b * S_ + s) * NH + h) * HD_ + d;
  float v = src[si];
  red[d] = v * v;
  __syncthreads();
  #pragma unroll
  for (int off = 64; off > 0; off >>= 1) {
    if (d < off) red[d] += red[d + off];
    __syncthreads();
  }
  const float rinv = rsqrtf(red[0] * (1.0f / HD_) + EPS_);
  nv[d] = v * rinv;
  __syncthreads();

  if (d < 64) {
    const float fr = (float)s * __expf(-(float)d * LN1E4_OVER_64);
    const float c  = __cosf(fr);
    const float sn = __sinf(fr);
    const float g  = (gain ? gain[h] : 1.0f) * scale;
    const float t1 = nv[d], t2 = nv[d + 64];
    const size_t o = (((size_t)b * NH + h) * S_ + s) * HD_;
    dst[o + d]      = to_bf16((t1 * c + t2 * sn) * g);
    dst[o + d + 64] = to_bf16((-t1 * sn + t2 * c) * g);
  }
}

// ---------------------------------------------------------------------------
// V: fp32 [B,S,KVH,HD] -> bf16 [B,KVH,HD,S]  (d-major so the P*V operand
// fragment is a contiguous LDS read in the attention kernel)
// ---------------------------------------------------------------------------
__global__ __launch_bounds__(256) void cvt_v(const float* __restrict__ vf,
                                             __bf16* __restrict__ vb)
{
  size_t i = (size_t)blockIdx.x * 256 + threadIdx.x;  // over B*S*KVD
  int d = (int)(i % HD_);
  size_t t = i / HD_;
  int h = (int)(t % KVH_); t /= KVH_;
  int s = (int)(t % S_);
  int b = (int)(t / S_);
  vb[(((size_t)b * KVH_ + h) * HD_ + d) * S_ + s] = to_bf16(vf[i]);
}

// ---------------------------------------------------------------------------
// Causal flash attention, GQA. One block per (64-query tile, h, b).
// Tile staging uses global_load_async_to_lds_b128 (ASYNCcnt); V's copies
// overlap the softmax. S = Q K^T and O += P V via v_wmma_f32_16x16x32_bf16;
// all operand fragments are contiguous 32B LDS loads.
// ---------------------------------------------------------------------------
__global__ __launch_bounds__(256) void flash_attn(
    const __bf16* __restrict__ Qb,   // [B,H,S,HD]
    const __bf16* __restrict__ Kb,   // [B,KVH,S,HD]
    const __bf16* __restrict__ Vb,   // [B,KVH,HD,S]
    __bf16* __restrict__ Ob)         // [B,S,H,HD]
{
  const int qt = blockIdx.x;
  const int h  = blockIdx.y;
  const int b  = blockIdx.z;
  const int kvh = h / (H_ / KVH_);
  const int tid = threadIdx.x;
  const int w    = tid >> 5;
  const int lane = tid & 31;

  __shared__ __bf16 Qs[64][HD_];     // 16 KB, natural K order
  __shared__ __bf16 KVs[64 * HD_];   // 16 KB; K phase [n][d], V phase [d][n]
  __shared__ float  Ss[64][65];      // scores
  __shared__ __bf16 Ps[64][64];      // probabilities, natural columns
  __shared__ float  m_run[64], l_run[64], alpha_s[64];

  const __bf16* Qg = Qb + (((size_t)b * H_  + h)   * S_ + (size_t)qt * 64) * HD_;
  const __bf16* Kg = Kb +  ((size_t)b * KVH_ + kvh) * S_ * HD_;
  const __bf16* Vg = Vb +  ((size_t)b * KVH_ + kvh) * HD_ * S_;

  // ---- stage Q tile (async): destination group natural ------------
  for (int ch = tid; ch < 64 * (HD_ / 8); ch += 256) {
    const int row = ch >> 4;           // 16 chunks per row
    const int g   = ch & 15;
    async_ld_b128(&Qs[row][g * 8], Qg + (size_t)row * HD_ + g * 8);
  }
  if (tid < 64) { m_run[tid] = -1e30f; l_run[tid] = 0.0f; }

  const int wm = (w & 3) * 16;     // wave's M fragment
  const int wd = (w >> 2) * 64;    // wave's D half (4 fragments)
  const int wn = (w >> 2) * 32;    // wave's N pair in the 64x64 S tile
  const int lrow  = lane & 15;
  const int halfo = (lane >> 4) << 4;
  v8f acc[4] = {};
  wait_async0();
  __syncthreads();

  for (int j = 0; j <= qt; ++j) {
    // ---- K tile -> LDS [n][d] (async) ------------------------------------
    const __bf16* Kt = Kg + (size_t)j * 64 * HD_;
    for (int ch = tid; ch < 64 * (HD_ / 8); ch += 256) {
      const int row = ch >> 4, g = ch & 15;
      async_ld_b128(&KVs[row * HD_ + g * 8], Kt + (size_t)row * HD_ + g * 8);
    }
    wait_async0();
    __syncthreads();

    // ---- S = Q K^T (2 fragments per wave, HD in 4 WMMA k-steps) ----------
    {
      v8f s0 = {}, s1 = {};
      #pragma unroll
      for (int kk = 0; kk < HD_; kk += 32) {
        const v16bf af = ld_frag(&Qs[wm + lrow][kk + halfo]);
        const v16bf b0 = ld_frag(&KVs[(wn + lrow) * HD_ + kk + halfo]);
        const v16bf b1 = ld_frag(&KVs[(wn + 16 + lrow) * HD_ + kk + halfo]);
        s0 = wmma_bf16(af, b0, s0);
        s1 = wmma_bf16(af, b1, s1);
      }
      const int col = lane & 15, rb = (lane >> 4) << 3;
      #pragma unroll
      for (int r = 0; r < 8; ++r) {
        Ss[wm + rb + r][wn + col]      = s0[r];
        Ss[wm + rb + r][wn + 16 + col] = s1[r];
      }
    }
    __syncthreads();

    // ---- V tile -> LDS [d][n] (async, overlaps softmax) ------------------
    const __bf16* Vt = Vg + (size_t)j * 64;
    for (int ch = tid; ch < HD_ * 8; ch += 256) {       // 128 rows x 8 chunks
      const int d = ch >> 3, g = ch & 7;
      async_ld_b128(&KVs[d * 64 + g * 8], Vt + (size_t)d * S_ + g * 8);
    }

    if (tid < 64) {
      const int r  = tid;
      const int qi = qt * 64 + r;
      int lim = qi - j * 64 + 1;                 // causal valid keys
      if (lim > 64) lim = 64;
      float mx = m_run[r];
      for (int c = 0; c < lim; ++c) mx = fmaxf(mx, Ss[r][c]);
      const float al = __expf(m_run[r] - mx);
      float l = l_run[r] * al;
      for (int c = 0; c < 64; ++c) {
        float p = (c < lim) ? __expf(Ss[r][c] - mx) : 0.0f;
        Ps[r][c] = to_bf16(p);                   // natural A storage
        l += p;
      }
      m_run[r] = mx; l_run[r] = l; alpha_s[r] = al;
    }
    wait_async0();
    __syncthreads();

    // ---- rescale accumulators, O += P V ----------------------------------
    {
      const int rb = (lane >> 4) << 3;
      float alv[8];
      #pragma unroll
      for (int r = 0; r < 8; ++r) alv[r] = alpha_s[wm + rb + r];
      #pragma unroll
      for (int f = 0; f < 4; ++f)
        #pragma unroll
        for (int r = 0; r < 8; ++r) acc[f][r] *= alv[r];

      #pragma unroll
      for (int kk = 0; kk < 64; kk += 32) {
        const v16bf af = ld_frag(&Ps[wm + lrow][kk + halfo]);
        #pragma unroll
        for (int f = 0; f < 4; ++f) {
          const v16bf bf = ld_frag(&KVs[(wd + f * 16 + lrow) * 64 + kk + halfo]);
          acc[f] = wmma_bf16(af, bf, acc[f]);
        }
      }
    }
    __syncthreads();
  }

  // ---- epilogue: /= l, store bf16 [B,S,H,HD] ------------------------------
  const int col = lane & 15, rb = (lane >> 4) << 3;
  #pragma unroll
  for (int r = 0; r < 8; ++r) {
    const int m = wm + rb + r;
    const float linv = 1.0f / l_run[m];
    const int qi = qt * 64 + m;
    #pragma unroll
    for (int f = 0; f < 4; ++f) {
      const int d = wd + f * 16 + col;
      Ob[(((size_t)b * S_ + qi) * H_ + h) * HD_ + d] = to_bf16(acc[f][r] * linv);
    }
  }
}

// ---------------------------------------------------------------------------
// Host-side orchestration
// ---------------------------------------------------------------------------
extern "C" void kernel_launch(void* const* d_in, const int* in_sizes, int n_in,
                              void* d_out, int out_size, void* d_ws, size_t ws_size,
                              hipStream_t stream)
{
  (void)in_sizes; (void)n_in; (void)out_size; (void)ws_size;
  const float* x      = (const float*)d_in[0];
  const float* Wq     = (const float*)d_in[1];
  const float* bq     = (const float*)d_in[2];
  const float* Wk     = (const float*)d_in[3];
  const float* bk     = (const float*)d_in[4];
  const float* Wv     = (const float*)d_in[5];
  const float* bv     = (const float*)d_in[6];
  const float* Wo     = (const float*)d_in[7];
  const float* bo     = (const float*)d_in[8];
  const float* q_gain = (const float*)d_in[9];
  float* out = (float*)d_out;

  size_t off = 0;
  auto carve = [&](size_t bytes) {
    void* p = (char*)d_ws + off;
    off += (bytes + 255) & ~(size_t)255;
    return p;
  };
  float*  qf = (float*)carve((size_t)B_ * S_ * D_   * 4);   // [B,S,H,HD] fp32
  float*  kf = (float*)carve((size_t)B_ * S_ * KVD_ * 4);   // [B,S,KVH,HD] fp32
  float*  vf = (float*)carve((size_t)B_ * S_ * KVD_ * 4);
  __bf16* qb = (__bf16*)carve((size_t)B_ * S_ * D_   * 2);  // [B,H,S,HD] bf16
  __bf16* kb2= (__bf16*)carve((size_t)B_ * S_ * KVD_ * 2);  // [B,KVH,S,HD] bf16
  __bf16* vb = (__bf16*)carve((size_t)B_ * S_ * KVD_ * 2);  // [B,KVH,HD,S] bf16
  __bf16* ob = (__bf16*)carve((size_t)B_ * S_ * D_   * 2);  // [B,S,D] bf16

  const int M = B_ * S_;   // 4096

  // 1) QKV projections
  gemm_abt_wmma<float><<<dim3(D_ / 64,   M / 64), 256, 0, stream>>>(x, D_, Wq, D_, bq, qf, D_,   D_);
  gemm_abt_wmma<float><<<dim3(KVD_ / 64, M / 64), 256, 0, stream>>>(x, D_, Wk, D_, bk, kf, KVD_, D_);
  gemm_abt_wmma<float><<<dim3(KVD_ / 64, M / 64), 256, 0, stream>>>(x, D_, Wv, D_, bv, vf, KVD_, D_);

  // 2) RMSNorm + RoPE (+gain*HD^-0.5 folded into q)
  const float scale = 1.0f / sqrtf((float)HD_);
  rmsnorm_rope<<<dim3(S_, H_,   B_), 128, 0, stream>>>(qf, qb,  q_gain, scale);
  rmsnorm_rope<<<dim3(S_, KVH_, B_), 128, 0, stream>>>(kf, kb2, nullptr, 1.0f);

  // 3) V -> bf16, d-major
  cvt_v<<<(B_ * S_ * KVD_) / 256, 256, 0, stream>>>(vf, vb);

  // 4) Causal flash attention (GQA)
  flash_attn<<<dim3(S_ / 64, H_, B_), 256, 0, stream>>>(qb, kb2, vb, ob);

  // 5) Output projection
  gemm_abt_wmma<__bf16><<<dim3(D_ / 64, M / 64), 256, 0, stream>>>(ob, D_, Wo, D_, bo, out, D_, D_);
}
